// HLI_5463198400803
// MI455X (gfx1250) — matrix operand order repaired and checked by reading, not compiled, not run on hardware
//
#include <hip/hip_runtime.h>
#include <hip/hip_bf16.h>

// ---------------------------------------------------------------------------
// Dims: B=8 T=32 P=32 N=16 D=128, H=4 C=16 HD=32, FFN_PCI=1024, MLP_H=512.
// All GEMMs via v_wmma_f32_16x16x32_bf16. Activations live in LDS as bf16 so
// an A fragment is two ds_load_b128; weights are pre-packed bf16 B fragments
// (two global_load_b128). Wave mapping: one N tile per wave, all M tiles per
// wave -> each B fragment is loaded exactly once per block.
// ---------------------------------------------------------------------------

typedef __attribute__((ext_vector_type(16))) __bf16 v16bf;
typedef __attribute__((ext_vector_type(8)))  float  v8f;

union AB16 { v16bf v; unsigned int u[8]; uint4 q[2]; };

__device__ __forceinline__ unsigned int f2bf(float f) {
    unsigned int x = __float_as_uint(f);
    unsigned int r = x + 0x7FFFu + ((x >> 16) & 1u);   // RNE f32 -> bf16
    return r >> 16;
}
__device__ __forceinline__ unsigned int pk2bf(float lo, float hi) {
    return f2bf(lo) | (f2bf(hi) << 16);
}
__device__ __forceinline__ float bf2f(unsigned short h) {
    return __uint_as_float(((unsigned int)h) << 16);
}
__device__ __forceinline__ float gelu_f(float x) {
    return 0.5f * x * (1.f + erff(x * 0.7071067811865475f));
}

__device__ __forceinline__ v8f wmma_bf16(v16bf a, v16bf b, v8f c) {
    return __builtin_amdgcn_wmma_f32_16x16x32_bf16(
        false, a, false, b, (short)0, c, false, false);
}

// A fragment (16x32) from a bf16 row-major LDS tile: two ds_load_b128 per lane.
__device__ __forceinline__ v16bf afrag_bf(const unsigned short* A, int ldw,
                                          int mbase, int k0, int lane) {
    AB16 r;
    const uint4* p = (const uint4*)(A + (mbase + (lane & 15)) * ldw + k0 + ((lane >> 4) << 3));
    r.q[0] = p[0];   // K pairs {0..7}   for this lane half
    r.q[1] = p[2];   // K pairs {16..23} for this lane half
    return r.v;
}

// B fragment (32x16) from pre-packed bf16 weights: two global_load_b128 per lane.
__device__ __forceinline__ v16bf bfrag(const unsigned short* w, int tile, int lane) {
    AB16 r;
    const uint4* p = (const uint4*)(w + ((size_t)tile << 9) + (lane << 4));
    r.q[0] = p[0];
    r.q[1] = p[1];
    return r.v;
}

// ---------------------------------------------------------------------------
// Weight pre-pack: f32 [K,N] row-major -> bf16 WMMA B-fragment tiles.
// ---------------------------------------------------------------------------
__global__ __launch_bounds__(256) void pack_w_bf16(const float* __restrict__ src,
                                                   unsigned short* __restrict__ dst,
                                                   int K, int N, int total) {
    int i = blockIdx.x * 256 + threadIdx.x;
    if (i >= total) return;
    int tile = i >> 9, within = i & 511;
    int lane = within >> 4, e = within & 15;
    int ntiles = N >> 4;
    int kt = tile / ntiles, nt = tile - kt * ntiles;
    int k = (kt << 5) + ((lane >> 4) << 3) + (e & 7) + ((e >> 3) << 4);
    int n = (nt << 4) + (lane & 15);
    dst[i] = (unsigned short)f2bf(src[k * N + n]);
}

// ---------------------------------------------------------------------------
// Kernel A: fused PatchCoreInteraction (inter stage), one block per (b,t,n).
// ---------------------------------------------------------------------------
#define XLDW 136   // bf16 row stride (pad 8)
#define FLDW 264   // bf16 row stride for 256-wide concat buffer
#define QLDW 132   // f32 row stride

__global__ __launch_bounds__(256) void pci_inter_kernel(
    const float* __restrict__ x, const float* __restrict__ cores,
    const unsigned short* __restrict__ vw_pk, const float* __restrict__ vb,
    const unsigned short* __restrict__ w1_pk, const float* __restrict__ b1,
    const unsigned short* __restrict__ w2_pk, const float* __restrict__ b2,
    const float* __restrict__ ng, const float* __restrict__ nb,
    float* __restrict__ xout) {
    __shared__ __align__(16) unsigned char spool[50176];            // 49 KB
    unsigned short* s_xnb = (unsigned short*)spool;                 // [32][136] bf16 LN(x)
    float*          s_q   = (float*)(spool + 8704);                 // [32][132] q -> vout -> y
    unsigned short* s_hb  = (unsigned short*)(spool + 8704);        // [32][136] hidden (overlay)
    float*          s_aff = (float*)(spool + 25600);                // [4][16][32]
    float*          s_c2p = (float*)(spool + 25600 + 8192);         // [4][16][32]
    float*          s_vc  = (float*)(spool + 25600 + 16384);        // [4][16][32]
    unsigned short* s_fb  = (unsigned short*)(spool + 25600);       // [32][264] ffn_in (overlay)

    const int tid  = threadIdx.x;
    const int lane = tid & 31;
    const int wv   = tid >> 5;                   // wave's N tile (0..7)
    const int m    = blockIdx.x;
    const int bt   = m >> 4;
    const int nn   = m & 15;

    // ---- Phase 1: load + LN (eps 1e-6) -> bf16 LDS ----
    {
        int p = tid >> 3, lig = tid & 7;
        const float* rp = x + ((size_t)(bt * 32 + p) * 16 + nn) * 128 + lig * 16;
        float v[16]; float s = 0.f, ss = 0.f;
#pragma unroll
        for (int j = 0; j < 16; ++j) { v[j] = rp[j]; s += v[j]; ss += v[j] * v[j]; }
#pragma unroll
        for (int off = 1; off < 8; off <<= 1) {
            s += __shfl_xor(s, off, 32);
            ss += __shfl_xor(ss, off, 32);
        }
        float mu = s * (1.f / 128.f);
        float rs = rsqrtf(ss * (1.f / 128.f) - mu * mu + 1e-6f);
        unsigned int* wp = (unsigned int*)(s_xnb + p * XLDW + lig * 16);
#pragma unroll
        for (int j = 0; j < 8; ++j)
            wp[j] = pk2bf((v[2 * j] - mu) * rs, (v[2 * j + 1] - mu) * rs);
    }
    __syncthreads();

    const int col    = (wv << 4) + (lane & 15);  // wave's output column
    const int rowoff = (lane < 16) ? 0 : 8;

    // ---- Phase 2: q = xn @ vw + vb  (wave: nt=wv, mt=0..1; B loaded once) ----
    {
        v8f c0 = {}, c1 = {};
        for (int kt = 0; kt < 4; ++kt) {
            v16bf b  = bfrag(vw_pk, kt * 8 + wv, lane);
            v16bf a0 = afrag_bf(s_xnb, XLDW, 0,  kt * 32, lane);
            v16bf a1 = afrag_bf(s_xnb, XLDW, 16, kt * 32, lane);
            c0 = wmma_bf16(a0, b, c0);
            c1 = wmma_bf16(a1, b, c1);
        }
        float bb = vb[col];
#pragma unroll
        for (int r = 0; r < 8; ++r) {
            s_q[(rowoff + r) * QLDW + col]      = c0[r] + bb;
            s_q[(16 + rowoff + r) * QLDW + col] = c1[r] + bb;
        }
    }
    __syncthreads();

    // ---- Phase 3: aff[h][c][p] = cores[h,c,:] . q[p, h*32:] / sqrt(32) ----
#pragma unroll
    for (int i = 0; i < 8; ++i) {
        int idx = tid + (i << 8);
        int h = idx >> 9, c = (idx >> 5) & 15, p = idx & 31;
        const float* cp = cores + (h * 16 + c) * 32;
        const float* qp = s_q + p * QLDW + h * 32;
        float acc = 0.f;
#pragma unroll
        for (int d = 0; d < 32; ++d) acc += cp[d] * qp[d];
        s_aff[idx] = acc * 0.17677669529663687f;
    }
    __syncthreads();

    // ---- Phase 4a: a_c2p = softmax over p ----
    if (tid < 64) {
        const float* ap = s_aff + tid * 32;
        float mx = -1e30f;
        for (int p = 0; p < 32; ++p) mx = fmaxf(mx, ap[p]);
        float t[32], sum = 0.f;
        for (int p = 0; p < 32; ++p) { t[p] = __expf(ap[p] - mx); sum += t[p]; }
        float inv = 1.f / sum;
        for (int p = 0; p < 32; ++p) s_c2p[tid * 32 + p] = t[p] * inv;
    }
    __syncthreads();
    // ---- Phase 4b: a_p2c = softmax over c (in place on aff) ----
    if (tid < 128) {
        int h = tid >> 5, p = tid & 31;
        float* base = s_aff + h * 512 + p;
        float mx = -1e30f;
        for (int c = 0; c < 16; ++c) mx = fmaxf(mx, base[c * 32]);
        float t[16], sum = 0.f;
        for (int c = 0; c < 16; ++c) { t[c] = __expf(base[c * 32] - mx); sum += t[c]; }
        float inv = 1.f / sum;
        for (int c = 0; c < 16; ++c) base[c * 32] = t[c] * inv;
    }
    __syncthreads();

    // ---- Phase 5: v_core[h][c][d] = sum_p v[p][h*32+d] * c2p[h][c][p] ----
#pragma unroll
    for (int i = 0; i < 8; ++i) {
        int idx = tid + (i << 8);
        int h = idx >> 9, c = (idx >> 5) & 15, d = idx & 31;
        const float* w = s_c2p + (h * 16 + c) * 32;
        const unsigned short* vcol = s_xnb + h * 32 + d;
        float acc = 0.f;
#pragma unroll
        for (int p = 0; p < 32; ++p) acc += w[p] * bf2f(vcol[p * XLDW]);
        s_vc[idx] = acc;
    }
    __syncthreads();

    // ---- Phase 6: v_out[p][h*32+d] = sum_c v_core[h][c][d] * p2c[h][c][p] -> s_q ----
#pragma unroll
    for (int i = 0; i < 16; ++i) {
        int idx = tid + (i << 8);
        int h = idx >> 10, p = (idx >> 5) & 31, d = idx & 31;
        const float* vc = s_vc + h * 512 + d;
        const float* a  = s_aff + h * 512 + p;
        float acc = 0.f;
#pragma unroll
        for (int c = 0; c < 16; ++c) acc += vc[c * 32] * a[c * 32];
        s_q[p * QLDW + h * 32 + d] = acc;
    }
    __syncthreads();

    // ---- Phase 7a: ffn_in = [xn - vout, vout] as bf16 (overlay attn scratch) ----
#pragma unroll
    for (int i = 0; i < 32; ++i) {       // row = i, col = tid
        int cc = tid;
        float val;
        if (cc < 128) val = bf2f(s_xnb[i * XLDW + cc]) - s_q[i * QLDW + cc];
        else          val = s_q[i * QLDW + cc - 128];
        s_fb[i * FLDW + cc] = (unsigned short)f2bf(val);
    }
    __syncthreads();

    // ---- Phase 7b: FFN 256 -> 1024 -> 128, hidden streamed in 8 chunks ----
    v8f co0 = {}, co1 = {};
    for (int ch = 0; ch < 8; ++ch) {
        __builtin_prefetch((const void*)(w1_pk + ((size_t)(ch * 8 + wv) << 9)), 0, 1);
        v8f h0 = {}, h1 = {};
        for (int kt = 0; kt < 8; ++kt) {                 // K = 256
            v16bf b  = bfrag(w1_pk, kt * 64 + ch * 8 + wv, lane);
            v16bf a0 = afrag_bf(s_fb, FLDW, 0,  kt * 32, lane);
            v16bf a1 = afrag_bf(s_fb, FLDW, 16, kt * 32, lane);
            h0 = wmma_bf16(a0, b, h0);
            h1 = wmma_bf16(a1, b, h1);
        }
        {   // gelu(h + b1) -> bf16 hidden tile
            float bb = b1[ch * 128 + col];
#pragma unroll
            for (int r = 0; r < 8; ++r) {
                s_hb[(rowoff + r) * XLDW + col]      = (unsigned short)f2bf(gelu_f(h0[r] + bb));
                s_hb[(16 + rowoff + r) * XLDW + col] = (unsigned short)f2bf(gelu_f(h1[r] + bb));
            }
        }
        __syncthreads();
        for (int kt = 0; kt < 4; ++kt) {                 // partial K of ffn2
            v16bf b  = bfrag(w2_pk, (ch * 4 + kt) * 8 + wv, lane);
            v16bf a0 = afrag_bf(s_hb, XLDW, 0,  kt * 32, lane);
            v16bf a1 = afrag_bf(s_hb, XLDW, 16, kt * 32, lane);
            co0 = wmma_bf16(a0, b, co0);
            co1 = wmma_bf16(a1, b, co1);
        }
        __syncthreads();
    }

    // ---- Phase 8: y = xn + ffn_out + b2 -> s_q ----
    {
        float bb = b2[col];
#pragma unroll
        for (int r = 0; r < 8; ++r) {
            int r0 = rowoff + r, r1 = 16 + rowoff + r;
            s_q[r0 * QLDW + col] = bf2f(s_xnb[r0 * XLDW + col]) + co0[r] + bb;
            s_q[r1 * QLDW + col] = bf2f(s_xnb[r1 * XLDW + col]) + co1[r] + bb;
        }
    }
    __syncthreads();

    // ---- Phase 9: affine LN (eps 1e-5) + global residual, store ----
    {
        int p = tid >> 3, lig = tid & 7;
        const float* yr = s_q + p * QLDW + lig * 16;
        float v[16]; float s = 0.f, ss = 0.f;
#pragma unroll
        for (int j = 0; j < 16; ++j) { v[j] = yr[j]; s += v[j]; ss += v[j] * v[j]; }
#pragma unroll
        for (int off = 1; off < 8; off <<= 1) {
            s += __shfl_xor(s, off, 32);
            ss += __shfl_xor(ss, off, 32);
        }
        float mu = s * (1.f / 128.f);
        float rs = rsqrtf(ss * (1.f / 128.f) - mu * mu + 1e-5f);
        size_t base = ((size_t)(bt * 32 + p) * 16 + nn) * 128 + lig * 16;
        const float* xr = x + base;
        float* orow = xout + base;
#pragma unroll
        for (int j = 0; j < 16; ++j) {
            int d = lig * 16 + j;
            orow[j] = xr[j] + ((v[j] - mu) * rs * ng[d] + nb[d]);
        }
    }
}

// ---------------------------------------------------------------------------
// Kernels B/C/D: out = in + gelu(LN(in,1e-6) @ W1 + b1) @ W2 + b2
// M=64 rows per block (4 m-tiles), one N tile per wave, all-bf16 LDS (35 KB).
// ---------------------------------------------------------------------------
template <int HID>
__global__ __launch_bounds__(256) void mlp_res_kernel(
    const float* __restrict__ in,
    const unsigned short* __restrict__ w1_pk, const float* __restrict__ b1,
    const unsigned short* __restrict__ w2_pk, const float* __restrict__ b2,
    float* __restrict__ out) {
    __shared__ __align__(16) unsigned short s_xnb[64 * XLDW];
    __shared__ __align__(16) unsigned short s_hb[64 * XLDW];
    const int tid = threadIdx.x, lane = tid & 31, wv = tid >> 5;
    const size_t row0 = (size_t)blockIdx.x * 64;

    {   // LN (eps 1e-6) -> bf16 LDS; 4 threads per row, 32 cols each
        int p = tid >> 2, lig = tid & 3;
        const float* rp = in + (row0 + p) * 128 + lig * 32;
        float v[32]; float s = 0.f, ss = 0.f;
#pragma unroll
        for (int j = 0; j < 32; ++j) { v[j] = rp[j]; s += v[j]; ss += v[j] * v[j]; }
#pragma unroll
        for (int off = 1; off < 4; off <<= 1) {
            s += __shfl_xor(s, off, 32);
            ss += __shfl_xor(ss, off, 32);
        }
        float mu = s * (1.f / 128.f);
        float rs = rsqrtf(ss * (1.f / 128.f) - mu * mu + 1e-6f);
        unsigned int* wp = (unsigned int*)(s_xnb + p * XLDW + lig * 32);
#pragma unroll
        for (int j = 0; j < 16; ++j)
            wp[j] = pk2bf((v[2 * j] - mu) * rs, (v[2 * j + 1] - mu) * rs);
    }
    __syncthreads();

    const int col    = (wv << 4) + (lane & 15);
    const int rowoff = (lane < 16) ? 0 : 8;
    constexpr int NCH = HID / 128;
    constexpr int NT1 = HID / 16;
    v8f co[4] = {};

    for (int ch = 0; ch < NCH; ++ch) {
        __builtin_prefetch((const void*)(w1_pk + ((size_t)(ch * 8 + wv) << 9)), 0, 1);
        v8f h[4] = {};
        for (int kt = 0; kt < 4; ++kt) {                 // K = 128
            v16bf b = bfrag(w1_pk, kt * NT1 + ch * 8 + wv, lane);
#pragma unroll
            for (int mt = 0; mt < 4; ++mt) {
                v16bf a = afrag_bf(s_xnb, XLDW, mt * 16, kt * 32, lane);
                h[mt] = wmma_bf16(a, b, h[mt]);
            }
        }
        {
            float bb = b1[ch * 128 + col];
#pragma unroll
            for (int mt = 0; mt < 4; ++mt)
#pragma unroll
                for (int r = 0; r < 8; ++r) {
                    int row = mt * 16 + rowoff + r;
                    s_hb[row * XLDW + col] = (unsigned short)f2bf(gelu_f(h[mt][r] + bb));
                }
        }
        __syncthreads();
        for (int kt = 0; kt < 4; ++kt) {
            v16bf b = bfrag(w2_pk, (ch * 4 + kt) * 8 + wv, lane);
#pragma unroll
            for (int mt = 0; mt < 4; ++mt) {
                v16bf a = afrag_bf(s_hb, XLDW, mt * 16, kt * 32, lane);
                co[mt] = wmma_bf16(a, b, co[mt]);
            }
        }
        __syncthreads();
    }

    {   // residual + bias, store
        float bb = b2[col];
#pragma unroll
        for (int mt = 0; mt < 4; ++mt)
#pragma unroll
            for (int r = 0; r < 8; ++r) {
                size_t row = row0 + mt * 16 + rowoff + r;
                out[row * 128 + col] = in[row * 128 + col] + co[mt][r] + bb;
            }
    }
}

// ---------------------------------------------------------------------------
// Host side
// ---------------------------------------------------------------------------
static const size_t OFF_VW  = 0;
static const size_t OFF_W1  = OFF_VW  + 128 * 128;
static const size_t OFF_W2  = OFF_W1  + 256 * 1024;
static const size_t OFF_BF1 = OFF_W2  + 1024 * 128;
static const size_t OFF_BF2 = OFF_BF1 + 128 * 512;
static const size_t OFF_IF1 = OFF_BF2 + 512 * 128;
static const size_t OFF_IF2 = OFF_IF1 + 128 * 128;
static const size_t OFF_MF1 = OFF_IF2 + 128 * 128;
static const size_t OFF_MF2 = OFF_MF1 + 128 * 512;
static const size_t WS_X1_BYTE_OFF = 2u << 20;   // packed weights < 2MB

extern "C" void kernel_launch(void* const* d_in, const int* in_sizes, int n_in,
                              void* d_out, int out_size, void* d_ws, size_t ws_size,
                              hipStream_t stream) {
    (void)in_sizes; (void)n_in; (void)out_size; (void)ws_size;
    const float* x    = (const float*)d_in[0];
    const float* cores= (const float*)d_in[1];
    const float* vw   = (const float*)d_in[2];
    const float* vb   = (const float*)d_in[3];
    const float* f1w  = (const float*)d_in[4];
    const float* f1b  = (const float*)d_in[5];
    const float* f2w  = (const float*)d_in[6];
    const float* f2b  = (const float*)d_in[7];
    const float* ng   = (const float*)d_in[8];
    const float* nb   = (const float*)d_in[9];
    const float* im1w = (const float*)d_in[10];
    const float* im1b = (const float*)d_in[11];
    const float* im2w = (const float*)d_in[12];
    const float* im2b = (const float*)d_in[13];
    const float* ii1w = (const float*)d_in[14];
    const float* ii1b = (const float*)d_in[15];
    const float* ii2w = (const float*)d_in[16];
    const float* ii2b = (const float*)d_in[17];
    const float* am1w = (const float*)d_in[18];
    const float* am1b = (const float*)d_in[19];
    const float* am2w = (const float*)d_in[20];
    const float* am2b = (const float*)d_in[21];

    unsigned short* wpk = (unsigned short*)d_ws;
    float* x1   = (float*)((char*)d_ws + WS_X1_BYTE_OFF);
    float* outp = (float*)d_out;

    struct PackJob { const float* src; size_t off; int K, N; };
    const PackJob jobs[9] = {
        { vw,   OFF_VW,  128,  128 }, { f1w,  OFF_W1,  256, 1024 },
        { f2w,  OFF_W2, 1024,  128 }, { im1w, OFF_BF1, 128,  512 },
        { im2w, OFF_BF2, 512,  128 }, { ii1w, OFF_IF1, 128,  128 },
        { ii2w, OFF_IF2, 128,  128 }, { am1w, OFF_MF1, 128,  512 },
        { am2w, OFF_MF2, 512,  128 },
    };
    for (int i = 0; i < 9; ++i) {
        int total = jobs[i].K * jobs[i].N;
        pack_w_bf16<<<(total + 255) / 256, 256, 0, stream>>>(
            jobs[i].src, wpk + jobs[i].off, jobs[i].K, jobs[i].N, total);
    }

    // 1) inter-patch PCI (fused):  x1 = x + inter
    pci_inter_kernel<<<4096, 256, 0, stream>>>(
        x, cores, wpk + OFF_VW, vb, wpk + OFF_W1, f1b, wpk + OFF_W2, f2b, ng, nb, x1);
    // 2) inter MLP:   out = x1 + mlp(ln(x1))
    mlp_res_kernel<512><<<2048, 256, 0, stream>>>(
        x1, wpk + OFF_BF1, im1b, wpk + OFF_BF2, im2b, outp);
    // 3) intra interaction (pointwise 128->128->128): x1 = out + mlp(ln(out))
    mlp_res_kernel<128><<<2048, 256, 0, stream>>>(
        outp, wpk + OFF_IF1, ii1b, wpk + OFF_IF2, ii2b, x1);
    // 4) intra MLP:   out = x1 + mlp(ln(x1))
    mlp_res_kernel<512><<<2048, 256, 0, stream>>>(
        x1, wpk + OFF_MF1, am1b, wpk + OFF_MF2, am2b, outp);
}
